// BiMambaEncoder_26319559590222
// MI455X (gfx1250) — compile-verified, matrix operand order checked by reading, not run on hardware
//
#include <hip/hip_runtime.h>
#include <hip/hip_bf16.h>

// ---------------------------------------------------------------------------
// BiMamba encoder for gfx1250 (MI455X).
//   - GEMMs (in-proj, x-proj, out-proj) use v_wmma_f32_16x16x32_f16, wave32.
//     Each wave computes a 32x64 tile (2 A fragments x 4 B fragments -> 8
//     WMMAs per 32-K step, each B fragment reused twice) with global_prefetch
//     of the next K chunk to hide load latency behind the WMMAs.
//   - Sequential SSM scan: 1 thread per (batch, channel), h-state in VGPRs,
//     per-timestep dt/B/C broadcast through LDS.
// ---------------------------------------------------------------------------

#define D_MODEL 256
#define D_INNER 512
#define D_STATE 16
#define DT_RANK 16
#define N_LAYERS 4
#define BB 8
#define LL 512
#define NROWS (BB * LL)       // 4096
#define EPSV 1e-5f

typedef __attribute__((ext_vector_type(16))) _Float16 v16h;
typedef __attribute__((ext_vector_type(8)))  _Float16 v8h;
typedef __attribute__((ext_vector_type(8)))  float    v8f;

// ---------------------------------------------------------------------------
// f32 -> f16 conversion (grid-stride)
// ---------------------------------------------------------------------------
__global__ void cvt_f16_kernel(const float* __restrict__ src,
                               _Float16* __restrict__ dst, int n) {
  int i = blockIdx.x * blockDim.x + threadIdx.x;
  int stride = gridDim.x * blockDim.x;
  for (; i < n; i += stride) dst[i] = (_Float16)src[i];
}

// ---------------------------------------------------------------------------
// Generic WMMA GEMM:  C[M,N] = A[M,K] * B[N,K]^T  (+ C if beta)
//   A, B in f16 row-major with K contiguous; C f32 row-major.
//   M % 32 == 0, K % 32 == 0.  Each wave computes one 32x64 tile:
//   2 M-subtiles x 4 N-subtiles, B fragments reused across both A fragments.
// ---------------------------------------------------------------------------
__global__ __launch_bounds__(256)
void wmma_gemm_kernel(const _Float16* __restrict__ A,
                      const _Float16* __restrict__ Bw,
                      float* __restrict__ C,
                      int M, int N, int K, int beta) {
  const int lane = threadIdx.x & 31;
  const int half = lane >> 4;          // 0: lanes 0-15, 1: lanes 16-31
  const int r    = lane & 15;
  const int wave = blockIdx.x * (blockDim.x >> 5) + (threadIdx.x >> 5);
  const int tilesN = (N + 63) >> 6;
  const int m0 = (wave / tilesN) * 32;
  if (m0 >= M) return;
  const int n0 = (wave % tilesN) * 64;

  v8f cacc[2][4] = {};
  if (beta) {
#pragma unroll
    for (int g = 0; g < 2; ++g) {
#pragma unroll
      for (int s = 0; s < 4; ++s) {
        int col = n0 + s * 16 + r;
        if (col < N) {
#pragma unroll
          for (int v = 0; v < 8; ++v) {
            int row = m0 + g * 16 + v + half * 8;
            cacc[g][s][v] = C[(size_t)row * N + col];
          }
        }
      }
    }
  }

  // A fragment rows for this lane (two M-subtiles), 4 B rows (clamped).
  const _Float16* arow0 = A + (size_t)(m0 + r) * K;
  const _Float16* arow1 = A + (size_t)(m0 + 16 + r) * K;
  const _Float16* brow[4];
#pragma unroll
  for (int s = 0; s < 4; ++s) {
    int nr = n0 + s * 16 + r;
    if (nr > N - 1) nr = N - 1;
    brow[s] = Bw + (size_t)nr * K;
  }

  for (int k0 = 0; k0 < K; k0 += 32) {
    if (k0 + 32 < K) {
      // Prefetch next K-chunk lines (lowers to global_prefetch_b8).
      __builtin_prefetch(arow0 + k0 + 32, 0, 1);
      __builtin_prefetch(arow1 + k0 + 32, 0, 1);
      __builtin_prefetch(brow[0] + k0 + 32, 0, 1);
      __builtin_prefetch(brow[2] + k0 + 32, 0, 1);
    }
    // 16-bit A 16x32 layout: lanes 0-15 hold K {0..7,16..23},
    // lanes 16-31 hold K {8..15,24..31} of this k-chunk.
    v8h a0lo = *(const v8h*)(arow0 + k0 + half * 8);
    v8h a0hi = *(const v8h*)(arow0 + k0 + 16 + half * 8);
    v16h a0 = __builtin_shufflevector(a0lo, a0hi, 0, 1, 2, 3, 4, 5, 6, 7,
                                      8, 9, 10, 11, 12, 13, 14, 15);
    v8h a1lo = *(const v8h*)(arow1 + k0 + half * 8);
    v8h a1hi = *(const v8h*)(arow1 + k0 + 16 + half * 8);
    v16h a1 = __builtin_shufflevector(a1lo, a1hi, 0, 1, 2, 3, 4, 5, 6, 7,
                                      8, 9, 10, 11, 12, 13, 14, 15);
#pragma unroll
    for (int s = 0; s < 4; ++s) {
      v8h blo = *(const v8h*)(brow[s] + k0 + half * 8);
      v8h bhi = *(const v8h*)(brow[s] + k0 + 16 + half * 8);
      v16h b = __builtin_shufflevector(blo, bhi, 0, 1, 2, 3, 4, 5, 6, 7,
                                       8, 9, 10, 11, 12, 13, 14, 15);
      cacc[0][s] = __builtin_amdgcn_wmma_f32_16x16x32_f16(
          false, a0, false, b, (short)0, cacc[0][s], false, false);
      cacc[1][s] = __builtin_amdgcn_wmma_f32_16x16x32_f16(
          false, a1, false, b, (short)0, cacc[1][s], false, false);
    }
  }

#pragma unroll
  for (int g = 0; g < 2; ++g) {
#pragma unroll
    for (int s = 0; s < 4; ++s) {
      int col = n0 + s * 16 + r;
      if (col < N) {
#pragma unroll
        for (int v = 0; v < 8; ++v) {
          int row = m0 + g * 16 + v + half * 8;
          C[(size_t)row * N + col] = cacc[g][s][v];
        }
      }
    }
  }
}

// ---------------------------------------------------------------------------
// Tokenizer: build 136-dim cat features, fuse, LayerNorm. 1 block per (b,l).
// ---------------------------------------------------------------------------
__global__ __launch_bounds__(256)
void tokenize_kernel(const float* __restrict__ x,
                     const float* __restrict__ ep, const float* __restrict__ ef,
                     const float* __restrict__ ed,
                     const float* __restrict__ Wlen, const float* __restrict__ blen,
                     const float* __restrict__ Wiat, const float* __restrict__ biat,
                     const float* __restrict__ Wfus, const float* __restrict__ bfus,
                     const float* __restrict__ g, const float* __restrict__ bb,
                     float* __restrict__ feat32, _Float16* __restrict__ feat16) {
  const int row = blockIdx.x;
  const int tid = threadIdx.x;
  __shared__ float cat[136];
  __shared__ float red[256];

  const float* xr = x + (size_t)row * 5;
  float len = xr[1], iat = xr[3];
  int pi = (int)xr[0]; pi = pi < 0 ? 0 : (pi > 255 ? 255 : pi);
  int fi = (int)xr[2]; fi = fi < 0 ? 0 : (fi > 63 ? 63 : fi);
  int di = (int)xr[4]; di = di < 0 ? 0 : (di > 1 ? 1 : di);

  if (tid < 32)        cat[tid] = ep[pi * 32 + tid];
  else if (tid < 64)   cat[tid] = len * Wlen[tid - 32] + blen[tid - 32];
  else if (tid < 96)   cat[tid] = ef[fi * 32 + (tid - 64)];
  else if (tid < 128)  cat[tid] = iat * Wiat[tid - 96] + biat[tid - 96];
  else if (tid < 136)  cat[tid] = ed[di * 8 + (tid - 128)];
  __syncthreads();

  float v = bfus[tid];
  const float* wr = Wfus + (size_t)tid * 136;
  for (int k = 0; k < 136; ++k) v += cat[k] * wr[k];

  // LayerNorm over 256 channels
  red[tid] = v; __syncthreads();
  for (int o = 128; o > 0; o >>= 1) {
    if (tid < o) red[tid] += red[tid + o];
    __syncthreads();
  }
  float mu = red[0] * (1.f / 256.f);
  __syncthreads();
  float dv = v - mu;
  red[tid] = dv * dv; __syncthreads();
  for (int o = 128; o > 0; o >>= 1) {
    if (tid < o) red[tid] += red[tid + o];
    __syncthreads();
  }
  float var = red[0] * (1.f / 256.f);
  float outv = dv * rsqrtf(var + EPSV) * g[tid] + bb[tid];
  feat32[(size_t)row * 256 + tid] = outv;
  feat16[(size_t)row * 256 + tid] = (_Float16)outv;
}

// ---------------------------------------------------------------------------
// Row LayerNorm (width 256): feat = LN(acc)*g + b, writes f32 + f16 shadow.
// ---------------------------------------------------------------------------
__global__ __launch_bounds__(256)
void rowln_kernel(const float* __restrict__ in,
                  const float* __restrict__ g, const float* __restrict__ bb,
                  float* __restrict__ out32, _Float16* __restrict__ out16) {
  const int row = blockIdx.x;
  const int tid = threadIdx.x;
  __shared__ float red[256];
  float v = in[(size_t)row * 256 + tid];
  red[tid] = v; __syncthreads();
  for (int o = 128; o > 0; o >>= 1) {
    if (tid < o) red[tid] += red[tid + o];
    __syncthreads();
  }
  float mu = red[0] * (1.f / 256.f);
  __syncthreads();
  float dv = v - mu;
  red[tid] = dv * dv; __syncthreads();
  for (int o = 128; o > 0; o >>= 1) {
    if (tid < o) red[tid] += red[tid + o];
    __syncthreads();
  }
  float var = red[0] * (1.f / 256.f);
  float outv = dv * rsqrtf(var + EPSV) * g[tid] + bb[tid];
  out32[(size_t)row * 256 + tid] = outv;
  out16[(size_t)row * 256 + tid] = (_Float16)outv;
}

// ---------------------------------------------------------------------------
// Depthwise causal conv (window 4) + bias + SiLU.
//   dir==0: out[l] = sum_j w[j]*x[l-3+j]   (causal)
//   dir==1: out[l] = sum_j w[j]*x[l+3-j]   (anti-causal == flip/conv/unflip)
// x is the first half of xz [.,.,1024]. Writes f32 + f16.
// ---------------------------------------------------------------------------
__global__ void conv_silu_kernel(const float* __restrict__ xz,
                                 const float* __restrict__ cw,
                                 const float* __restrict__ cb,
                                 float* __restrict__ x32,
                                 _Float16* __restrict__ x16, int dir) {
  int idx = blockIdx.x * blockDim.x + threadIdx.x;
  if (idx >= NROWS * D_INNER) return;
  int d = idx & (D_INNER - 1);
  int row = idx >> 9;
  int l = row & (LL - 1);
  int b = row >> 9;
  const float* base = xz + (size_t)b * LL * 1024 + d;
  const float* w = cw + d * 4;
  float acc = cb[d];
#pragma unroll
  for (int j = 0; j < 4; ++j) {
    int t = (dir == 0) ? (l - 3 + j) : (l + 3 - j);
    float xv = (t >= 0 && t < LL) ? base[(size_t)t * 1024] : 0.f;
    acc += w[j] * xv;
  }
  float o = acc / (1.f + __expf(-acc));   // SiLU
  x32[idx] = o;
  x16[idx] = (_Float16)o;
}

// ---------------------------------------------------------------------------
// Selective scan. 1 block per batch, 1 thread per channel d.
//   delta = softplus(dt . Wdt[d] + bdt[d])
//   h[s]  = exp(delta*A[d,s]) * h[s] + (delta*x) * B[s]
//   y     = (h . C + x*Dp[d]) * silu(z)
// dir==1 iterates time in reverse (flip/scan/unflip fused).
// ---------------------------------------------------------------------------
__global__ __launch_bounds__(512)
void scan_kernel(const float* __restrict__ xdbl,   // [B,L,48]
                 const float* __restrict__ x32,    // [B,L,512]
                 const float* __restrict__ xz,     // [B,L,1024] (z = [512:])
                 const float* __restrict__ dtw,    // [512,16]
                 const float* __restrict__ dtb,    // [512]
                 const float* __restrict__ Alog,   // [512,16]
                 const float* __restrict__ Dp,     // [512]
                 _Float16* __restrict__ y16,       // [B,L,512]
                 int dir) {
  const int b = blockIdx.x;
  const int d = threadIdx.x;
  float wdt[DT_RANK], a[D_STATE], h[D_STATE];
#pragma unroll
  for (int rr = 0; rr < DT_RANK; ++rr) wdt[rr] = dtw[d * DT_RANK + rr];
#pragma unroll
  for (int s = 0; s < D_STATE; ++s) {
    a[s] = -__expf(Alog[d * D_STATE + s]);
    h[s] = 0.f;
  }
  const float bd = dtb[d];
  const float Dd = Dp[d];
  __shared__ float s48[48];

  for (int step = 0; step < LL; ++step) {
    int t = (dir == 0) ? step : (LL - 1 - step);
    __syncthreads();
    if (d < 48) s48[d] = xdbl[((size_t)(b * LL + t)) * 48 + d];
    __syncthreads();

    float delta = bd;
#pragma unroll
    for (int rr = 0; rr < DT_RANK; ++rr) delta += s48[rr] * wdt[rr];
    delta = (delta > 20.f) ? delta : log1pf(__expf(delta));  // softplus

    const size_t ro = (size_t)(b * LL + t);
    float xv = x32[ro * D_INNER + d];
    float dx = delta * xv;
    float ys = 0.f;
#pragma unroll
    for (int s = 0; s < D_STATE; ++s) {
      float dA = __expf(delta * a[s]);
      h[s] = dA * h[s] + dx * s48[DT_RANK + s];
      ys += h[s] * s48[DT_RANK + D_STATE + s];
    }
    float y = ys + xv * Dd;
    float zv = xz[ro * 1024 + D_INNER + d];
    y *= zv / (1.f + __expf(-zv));        // * silu(z)
    y16[ro * D_INNER + d] = (_Float16)y;
  }
}

// ---------------------------------------------------------------------------
// Mean pool over L, then ReLU MLP head. 1 block per batch.
// ---------------------------------------------------------------------------
__global__ __launch_bounds__(256)
void pool_mlp_kernel(const float* __restrict__ feat,
                     const float* __restrict__ W1, const float* __restrict__ b1,
                     const float* __restrict__ W2, const float* __restrict__ b2,
                     float* __restrict__ out) {
  const int b = blockIdx.x;
  const int tid = threadIdx.x;
  __shared__ float pooled[256];
  __shared__ float hbuf[256];
  float s = 0.f;
  const float* fb = feat + (size_t)b * LL * 256 + tid;
  for (int l = 0; l < LL; ++l) s += fb[(size_t)l * 256];
  pooled[tid] = s * (1.f / (float)LL);
  __syncthreads();
  float hv = b1[tid];
  const float* w1 = W1 + (size_t)tid * 256;
  for (int c = 0; c < 256; ++c) hv += pooled[c] * w1[c];
  hbuf[tid] = hv > 0.f ? hv : 0.f;
  __syncthreads();
  float ov = b2[tid];
  const float* w2 = W2 + (size_t)tid * 256;
  for (int c = 0; c < 256; ++c) ov += hbuf[c] * w2[c];
  out[(size_t)b * 256 + tid] = ov;
}

// ---------------------------------------------------------------------------
// Host side
// ---------------------------------------------------------------------------
static inline void launch_gemm(const _Float16* A, const _Float16* Bw, float* C,
                               int M, int N, int K, int beta, hipStream_t s) {
  int tilesM = M / 32;                 // 32 rows per wave
  int tilesN = (N + 63) / 64;
  int waves = tilesM * tilesN;
  int blocks = (waves + 7) / 8;
  wmma_gemm_kernel<<<blocks, 256, 0, s>>>(A, Bw, C, M, N, K, beta);
}

static inline void launch_cvt(const float* src, _Float16* dst, int n,
                              hipStream_t s) {
  int blocks = (n + 255) / 256;
  if (blocks > 2048) blocks = 2048;
  cvt_f16_kernel<<<blocks, 256, 0, s>>>(src, dst, n);
}

extern "C" void kernel_launch(void* const* d_in, const int* in_sizes, int n_in,
                              void* d_out, int out_size, void* d_ws, size_t ws_size,
                              hipStream_t stream) {
  // --- inputs (setup_inputs dict order) ---
  const float* x        = (const float*)d_in[0];
  const float* emb_proto= (const float*)d_in[1];
  const float* emb_flags= (const float*)d_in[2];
  const float* emb_dir  = (const float*)d_in[3];
  const float* W_len    = (const float*)d_in[4];
  const float* b_len    = (const float*)d_in[5];
  const float* W_iat    = (const float*)d_in[6];
  const float* b_iat    = (const float*)d_in[7];
  const float* W_fus    = (const float*)d_in[8];
  const float* b_fus    = (const float*)d_in[9];
  const float* g_tok    = (const float*)d_in[10];
  const float* bt_tok   = (const float*)d_in[11];
  const float* g_norm   = (const float*)d_in[12];
  const float* b_norm   = (const float*)d_in[13];
  const float* W_h1     = (const float*)d_in[14];
  const float* b_h1     = (const float*)d_in[15];
  const float* W_h2     = (const float*)d_in[16];
  const float* b_h2     = (const float*)d_in[17];
  const float* in_w[2]   = {(const float*)d_in[18], (const float*)d_in[27]};
  const float* conv_w[2] = {(const float*)d_in[19], (const float*)d_in[28]};
  const float* conv_b[2] = {(const float*)d_in[20], (const float*)d_in[29]};
  const float* xproj[2]  = {(const float*)d_in[21], (const float*)d_in[30]};
  const float* dt_w[2]   = {(const float*)d_in[22], (const float*)d_in[31]};
  const float* dt_b[2]   = {(const float*)d_in[23], (const float*)d_in[32]};
  const float* Alog[2]   = {(const float*)d_in[24], (const float*)d_in[33]};
  const float* Dpar[2]   = {(const float*)d_in[25], (const float*)d_in[34]};
  const float* out_w[2]  = {(const float*)d_in[26], (const float*)d_in[35]};
  float* out = (float*)d_out;

  // --- workspace carve-up ---
  char* ws = (char*)d_ws;
  size_t off = 0;
  auto carve = [&](size_t bytes) -> char* {
    char* p = ws + off;
    off = (off + bytes + 255) & ~(size_t)255;
    return p;
  };
  float*    feat32 = (float*)   carve((size_t)NROWS * 256 * 4);
  _Float16* feat16 = (_Float16*)carve((size_t)NROWS * 256 * 2);
  float*    acc    = (float*)   carve((size_t)NROWS * 256 * 4);
  float*    xzbuf  = (float*)   carve((size_t)NROWS * 1024 * 4);
  float*    xbuf32 = (float*)   carve((size_t)NROWS * 512 * 4);
  _Float16* xbuf16 = (_Float16*)carve((size_t)NROWS * 512 * 2);
  float*    xdbl   = (float*)   carve((size_t)NROWS * 48 * 4);
  _Float16* ybuf16 = (_Float16*)carve((size_t)NROWS * 512 * 2);
  _Float16* wIn16[2], *wXp16[2], *wOut16[2];
  for (int dir = 0; dir < 2; ++dir) {
    wIn16[dir]  = (_Float16*)carve((size_t)N_LAYERS * 1024 * 256 * 2);
    wXp16[dir]  = (_Float16*)carve((size_t)N_LAYERS * 48 * 512 * 2);
    wOut16[dir] = (_Float16*)carve((size_t)N_LAYERS * 256 * 512 * 2);
  }

  // --- weight conversion to f16 (per launch; weights are small) ---
  for (int dir = 0; dir < 2; ++dir) {
    launch_cvt(in_w[dir],  wIn16[dir],  N_LAYERS * 1024 * 256, stream);
    launch_cvt(xproj[dir], wXp16[dir],  N_LAYERS * 48 * 512,   stream);
    launch_cvt(out_w[dir], wOut16[dir], N_LAYERS * 256 * 512,  stream);
  }

  // --- tokenizer ---
  tokenize_kernel<<<NROWS, 256, 0, stream>>>(
      x, emb_proto, emb_flags, emb_dir, W_len, b_len, W_iat, b_iat,
      W_fus, b_fus, g_tok, bt_tok, feat32, feat16);

  // --- layers ---
  for (int i = 0; i < N_LAYERS; ++i) {
    hipMemcpyAsync(acc, feat32, (size_t)NROWS * 256 * 4,
                   hipMemcpyDeviceToDevice, stream);   // residual init
    for (int dir = 0; dir < 2; ++dir) {
      const _Float16* wi = wIn16[dir]  + (size_t)i * 1024 * 256;
      const _Float16* wx = wXp16[dir]  + (size_t)i * 48 * 512;
      const _Float16* wo = wOut16[dir] + (size_t)i * 256 * 512;
      // in-proj: xz = feat @ Wp^T
      launch_gemm(feat16, wi, xzbuf, NROWS, 1024, 256, 0, stream);
      // depthwise conv + SiLU (direction-aware)
      conv_silu_kernel<<<(NROWS * 512 + 255) / 256, 256, 0, stream>>>(
          xzbuf, conv_w[dir] + (size_t)i * 512 * 4,
          conv_b[dir] + (size_t)i * 512, xbuf32, xbuf16, dir);
      // x-proj: xdbl = x @ Wx^T  (dt|B|C)
      launch_gemm(xbuf16, wx, xdbl, NROWS, 48, 512, 0, stream);
      // selective scan (+ delta, gating) -> y (f16)
      scan_kernel<<<BB, 512, 0, stream>>>(
          xdbl, xbuf32, xzbuf,
          dt_w[dir] + (size_t)i * 512 * 16, dt_b[dir] + (size_t)i * 512,
          Alog[dir] + (size_t)i * 512 * 16, Dpar[dir] + (size_t)i * 512,
          ybuf16, dir);
      // out-proj accumulated into residual: acc += y @ Wo^T
      launch_gemm(ybuf16, wo, acc, NROWS, 256, 512, 1, stream);
    }
    rowln_kernel<<<NROWS, 256, 0, stream>>>(acc, g_norm, b_norm, feat32, feat16);
  }

  // --- head ---
  pool_mlp_kernel<<<BB, 256, 0, stream>>>(feat32, W_h1, b_h1, W_h2, b_h2, out);
}
